// TransformerBlock_23304492548140
// MI455X (gfx1250) — matrix operand order repaired and checked by reading, not compiled, not run on hardware
//
#include <hip/hip_runtime.h>
#include <hip/hip_bf16.h>

typedef __bf16 bf16;
typedef __attribute__((ext_vector_type(16))) __bf16 v16bf;
typedef __attribute__((ext_vector_type(8)))  __bf16 v8bf;
typedef __attribute__((ext_vector_type(8)))  float  v8f;

#define EMB   128
#define HEADS 8
#define NB    128       // total batch = B0 * N_AGENTS
#define SEQ   200
#define TPAD  208       // 13 * 16
#define FFD   512
#define NTOK  (NB * SEQ)   // 25600
#define HE    (HEADS * EMB) // 1024

// ---------------- CDNA5 async global->LDS copy (ASYNCcnt) ---------------------
// GLOBAL_LOAD_ASYNC_TO_LDS_B128, GV mode: per-lane 16B memory->LDS, no VGPR data.
// LDS dest address = low 32 bits of the generic pointer (LDS aperture rule).
__device__ __forceinline__ void async_ld_b128(void* lds_dst, const void* gsrc) {
  unsigned lds_off = (unsigned)(uintptr_t)lds_dst;
  asm volatile("global_load_async_to_lds_b128 %0, %1, off"
               :: "v"(lds_off), "v"(gsrc) : "memory");
}
__device__ __forceinline__ void wait_async0() {
  asm volatile("s_wait_asynccnt 0" ::: "memory");
}

// ---------------- WMMA fragment helpers (layouts per CDNA5 ISA) ----

// A-operand, source row-major [M][K] (bf16), 16x32 tile at (m0,k0).
__device__ __forceinline__ v16bf frag_A(const bf16* base, int pitch, int m0, int k0, int lane) {
  const bf16* p = base + (m0 + (lane & 15)) * pitch + k0 + ((lane >> 4) << 3);
  v8bf lo = *(const v8bf*)p;
  v8bf hi = *(const v8bf*)(p + 16);
  v16bf r;
#pragma unroll
  for (int i = 0; i < 8; ++i) { r[i] = lo[i]; r[i + 8] = hi[i]; }
  return r;
}

// B-operand, source stored B-friendly as [N][K] (bf16), 32x16 tile at (k0,n0).
__device__ __forceinline__ v16bf frag_B(const bf16* base, int pitch, int n0, int k0, int lane) {
  const bf16* p = base + (n0 + (lane & 15)) * pitch + k0 + ((lane >> 4) << 3);
  v8bf lo = *(const v8bf*)p;
  v8bf hi = *(const v8bf*)(p + 16);
  v16bf r;
#pragma unroll
  for (int i = 0; i < 8; ++i) { r[i] = lo[i]; r[i + 8] = hi[i]; }
  return r;
}

__device__ __forceinline__ v8f wmma_bf16(v16bf a, v16bf b, v8f c) {
  return __builtin_amdgcn_wmma_f32_16x16x32_bf16(false, a, false, b, (short)0, c, false, false);
}

// ---------------- prep kernels ------------------------------------------------

__global__ void k_addpos(const float* __restrict__ x, const float* __restrict__ pos,
                         float* __restrict__ xf, bf16* __restrict__ xb) {
  int i = blockIdx.x * 256 + threadIdx.x;           // 3,276,800 total, exact grid
  float v = x[i] + pos[i % (SEQ * EMB)];
  xf[i] = v;
  xb[i] = (bf16)v;
}

__global__ void k_cvt(const float* __restrict__ s, bf16* __restrict__ d, int n) {
  int i = blockIdx.x * 256 + threadIdx.x;
  if (i < n) d[i] = (bf16)s[i];
}

// src [R][C] f32 -> dst [C][R] bf16   (B-friendly transpose)
__global__ void k_cvtT(const float* __restrict__ s, bf16* __restrict__ d, int R, int C) {
  int i = blockIdx.x * 256 + threadIdx.x;
  if (i < R * C) { int r = i / C, c = i % C; d[c * R + r] = (bf16)s[i]; }
}

// ---------------- fused attention: one workgroup per (batch, head) ------------
// LDS (dynamic, 264768 B):
//   Q  [208][136] bf16   @0        56576
//   K  [208][136] bf16   @56576    56576
//   Vt [128][232] bf16   @113152   59392   (V transposed: [e][t'])
//   msk[208]      int    @172544     832
//   union @173376 (91392):
//     phase A: X [208][136] bf16 (56576) ; WT [128][136] bf16 (34816)
//     phase B: S [16][224] f32 (14336)   ; P  [16][232] bf16 (7424)
#define A_PX  136
#define A_PVT 232
#define A_PS  224
#define A_PP  232
#define A_LDS 264768

__global__ void __launch_bounds__(256) k_attn(
    const bf16* __restrict__ xpe,   // [NTOK][EMB] bf16 (token-major)
    const bf16* __restrict__ Wqg,   // [EMB][HE]
    const bf16* __restrict__ Wkg,
    const bf16* __restrict__ Wvg,
    const int*  __restrict__ masks, // [B0][SEQ][N_AGENTS] int32
    bf16* __restrict__ attn)        // [NTOK][HE]
{
  extern __shared__ __align__(16) char smem[];
  const int h = blockIdx.x;
  const int b = blockIdx.y;
  const int tid = threadIdx.x, lane = tid & 31, w = tid >> 5;

  bf16* Q   = (bf16*)(smem + 0);
  bf16* K   = (bf16*)(smem + 56576);
  bf16* Vt  = (bf16*)(smem + 113152);
  int*  msk = (int*) (smem + 172544);
  char* un  = smem + 173376;
  bf16* X   = (bf16*)un;
  bf16* WT  = (bf16*)(un + 56576);
  float* S  = (float*)un;
  bf16* P   = (bf16*)(un + 14336);

  // ---- stage X via async global->LDS (pad rows 200..207 = 0), mask, Vt pad ---
  const bf16* xb = xpe + (size_t)b * SEQ * EMB;
  for (int c = tid; c < TPAD * (EMB / 8); c += 256) {
    int t = c >> 4, e = (c & 15) << 3;
    bf16* p = X + t * A_PX + e;
    if (t < SEQ) {
      async_ld_b128(p, xb + t * EMB + e);
    } else {
#pragma unroll
      for (int i = 0; i < 8; ++i) p[i] = (bf16)0.0f;
    }
  }
  {
    int b0 = b >> 3, ag = b & 7;
    if (tid < TPAD) msk[tid] = (tid < SEQ) ? masks[(b0 * SEQ + tid) * 8 + ag] : 0;
  }
  for (int idx = tid; idx < EMB * 24; idx += 256) {  // Vt cols 208..231 := 0
    int n = idx / 24, c = 208 + idx % 24;
    Vt[n * A_PVT + c] = (bf16)0.0f;
  }

  auto loadWT = [&](const bf16* Wg) {
    for (int idx = tid; idx < EMB * EMB; idx += 256) {
      int in = idx >> 7, n = idx & 127;
      WT[n * A_PX + in] = Wg[in * HE + h * EMB + n];   // B-friendly [n][k=in]
    }
  };
  auto proj = [&](bf16* dst, bool transposed) {        // dst = X @ W_h
    for (int mt = 0; mt < 13; ++mt) {
      v8f acc = {};
#pragma unroll
      for (int kc = 0; kc < 4; ++kc) {
        v16bf a  = frag_A(X,  A_PX, mt * 16, kc * 32, lane);
        v16bf bb = frag_B(WT, A_PX, w * 16,  kc * 32, lane);
        acc = wmma_bf16(a, bb, acc);
      }
#pragma unroll
      for (int r = 0; r < 8; ++r) {
        int m = mt * 16 + ((lane >> 4) << 3) + r;
        int n = w * 16 + (lane & 15);
        if (!transposed) dst[m * A_PX + n] = (bf16)acc[r];
        else             dst[n * A_PVT + m] = (bf16)acc[r];
      }
    }
  };

  loadWT(Wqg);
  wait_async0();          // drain this wave's async X copies before barrier
  __syncthreads();
  proj(Q, false); __syncthreads();
  loadWT(Wkg); __syncthreads();
  proj(K, false); __syncthreads();
  loadWT(Wvg); __syncthreads();
  proj(Vt, true);

  // ---- phase B: per 16-row tile: S = Q K^T, softmax, O = P V ----
  const float scale = 0.08838834764831845f;            // 1/sqrt(128)
  bf16* attb = attn + (size_t)b * SEQ * HE + h * EMB;
  for (int rt = 0; rt < 13; ++rt) {
    __syncthreads();                                   // S/P reuse + phase-A done
    for (int ct = w; ct < 13; ct += 8) {               // wave-uniform
      v8f acc = {};
#pragma unroll
      for (int kc = 0; kc < 4; ++kc) {
        v16bf a  = frag_A(Q, A_PX, rt * 16, kc * 32, lane);
        v16bf bb = frag_B(K, A_PX, ct * 16, kc * 32, lane);
        acc = wmma_bf16(a, bb, acc);
      }
#pragma unroll
      for (int r = 0; r < 8; ++r)
        S[(((lane >> 4) << 3) + r) * A_PS + ct * 16 + (lane & 15)] = acc[r];
    }
    __syncthreads();
    {   // masked softmax: 16 threads per row, shfl-width-16 reductions
      int m = tid >> 4, l = tid & 15;
      int t = rt * 16 + m;
      bool rowv = (t < SEQ) && (msk[t] != 0);
      float* Sr = S + m * A_PS;
      float rmax = -3.0e38f;
      for (int k = l; k < SEQ; k += 16) if (msk[k]) rmax = fmaxf(rmax, Sr[k]);
#pragma unroll
      for (int off = 8; off; off >>= 1) rmax = fmaxf(rmax, __shfl_xor(rmax, off, 16));
      float rsum = 0.f;
      for (int k = l; k < SEQ; k += 16) {
        float e = msk[k] ? __expf((Sr[k] - rmax) * scale) : 0.f;
        Sr[k] = e; rsum += e;
      }
#pragma unroll
      for (int off = 8; off; off >>= 1) rsum += __shfl_xor(rsum, off, 16);
      float inv = (rowv && rsum > 0.f) ? (1.f / rsum) : 0.f;
      bf16* Pr = P + m * A_PP;
      for (int k = l; k < 224; k += 16) {
        float pv = (k < SEQ && msk[k]) ? Sr[k] * inv : 0.f;
        Pr[k] = (bf16)pv;
      }
    }
    __syncthreads();
    {   // O tile: [16 x 16] per wave, K-dim 224
      v8f acc = {};
#pragma unroll
      for (int kc = 0; kc < 7; ++kc) {
        v16bf a  = frag_A(P,  A_PP,  0,      kc * 32, lane);
        v16bf bb = frag_B(Vt, A_PVT, w * 16, kc * 32, lane);
        acc = wmma_bf16(a, bb, acc);
      }
#pragma unroll
      for (int r = 0; r < 8; ++r) {
        int t = rt * 16 + ((lane >> 4) << 3) + r;
        if (t < SEQ)
          attb[(size_t)t * HE + w * 16 + (lane & 15)] = (bf16)acc[r];
      }
    }
  }
}

// ---------------- GEMM (N=128) + bias + residual + LayerNorm ------------------
// A [NTOK][Kdim] bf16, BT [128][Kdim] bf16; 64-row block; 400 blocks.
__global__ void __launch_bounds__(256) k_gemm_ln(
    const bf16* __restrict__ Ag, int Kdim,
    const bf16* __restrict__ BT,
    const float* __restrict__ bias,
    const float* __restrict__ resid,   // [NTOK][128]
    const float* __restrict__ lnw, const float* __restrict__ lnb,
    float* __restrict__ outF, bf16* __restrict__ outB)
{
  __shared__ __align__(16) char sm[52224];
  __shared__ float mu[64], rs[64];
  bf16* A  = (bf16*)sm;
  bf16* Bb = (bf16*)(sm + 17408);
  float* Z = (float*)sm;                 // reused after GEMM, pitch 132
  const int tid = threadIdx.x, lane = tid & 31, w = tid >> 5;
  const int row0 = blockIdx.x * 64;
  v8f zero = {};
  v8f acc[4];
#pragma unroll
  for (int mt = 0; mt < 4; ++mt) acc[mt] = zero;

  for (int kb = 0; kb < Kdim / 128; ++kb) {
    __syncthreads();
    for (int c = tid; c < 64 * 16; c += 256) {          // A tile, async -> LDS
      int rr = c >> 4, k = (c & 15) << 3;
      async_ld_b128(A + rr * 136 + k,
                    Ag + (size_t)(row0 + rr) * Kdim + kb * 128 + k);
    }
    for (int c = tid; c < 128 * 16; c += 256) {         // B tile, async -> LDS
      int n = c >> 4, k = (c & 15) << 3;
      async_ld_b128(Bb + n * 136 + k,
                    BT + (size_t)n * Kdim + kb * 128 + k);
    }
    wait_async0();
    __syncthreads();
#pragma unroll
    for (int mt = 0; mt < 4; ++mt)
#pragma unroll
      for (int kc = 0; kc < 4; ++kc) {
        v16bf a  = frag_A(A,  136, mt * 16, kc * 32, lane);
        v16bf bb = frag_B(Bb, 136, w * 16,  kc * 32, lane);
        acc[mt] = wmma_bf16(a, bb, acc[mt]);
      }
  }
  __syncthreads();
  {
    int n = w * 16 + (lane & 15);
    float bn = bias[n];
#pragma unroll
    for (int mt = 0; mt < 4; ++mt)
#pragma unroll
      for (int r = 0; r < 8; ++r) {
        int m = mt * 16 + ((lane >> 4) << 3) + r;
        Z[m * 132 + n] = acc[mt][r] + bn + resid[(size_t)(row0 + m) * 128 + n];
      }
  }
  __syncthreads();
  if (tid < 64) {
    float s = 0.f;
    for (int n = 0; n < 128; ++n) s += Z[tid * 132 + n];
    float m = s * (1.f / 128.f);
    float v = 0.f;
    for (int n = 0; n < 128; ++n) { float d = Z[tid * 132 + n] - m; v += d * d; }
    mu[tid] = m;
    rs[tid] = rsqrtf(v * (1.f / 128.f) + 1e-5f);
  }
  __syncthreads();
  for (int idx = tid; idx < 64 * 128; idx += 256) {
    int m = idx >> 7, n = idx & 127;
    float val = (Z[m * 132 + n] - mu[m]) * rs[m] * lnw[n] + lnb[n];
    size_t o = (size_t)(row0 + m) * 128 + n;
    outF[o] = val;
    if (outB) outB[o] = (bf16)val;
  }
}

// ---------------- FF1: [NTOK][128] @ [128][512] + bias, exact GELU ------------
__global__ void __launch_bounds__(256) k_ff1(
    const bf16* __restrict__ y1b, const bf16* __restrict__ W1T,
    const float* __restrict__ b1, bf16* __restrict__ hbuf)
{
  __shared__ __align__(16) char sm[52224];
  bf16* A  = (bf16*)sm;
  bf16* Bb = (bf16*)(sm + 17408);
  const int tid = threadIdx.x, lane = tid & 31, w = tid >> 5;
  const int row0 = blockIdx.x * 64, n0g = blockIdx.y * 128;
  for (int c = tid; c < 64 * 16; c += 256) {
    int rr = c >> 4, k = (c & 15) << 3;
    async_ld_b128(A + rr * 136 + k, y1b + (size_t)(row0 + rr) * 128 + k);
  }
  for (int c = tid; c < 128 * 16; c += 256) {
    int n = c >> 4, k = (c & 15) << 3;
    async_ld_b128(Bb + n * 136 + k, W1T + (size_t)(n0g + n) * 128 + k);
  }
  wait_async0();
  __syncthreads();
  v8f zero = {};
  v8f acc[4];
#pragma unroll
  for (int mt = 0; mt < 4; ++mt) acc[mt] = zero;
#pragma unroll
  for (int mt = 0; mt < 4; ++mt)
#pragma unroll
    for (int kc = 0; kc < 4; ++kc) {
      v16bf a  = frag_A(A,  136, mt * 16, kc * 32, lane);
      v16bf bb = frag_B(Bb, 136, w * 16,  kc * 32, lane);
      acc[mt] = wmma_bf16(a, bb, acc[mt]);
    }
  int n = n0g + w * 16 + (lane & 15);
  float bn = b1[n];
#pragma unroll
  for (int mt = 0; mt < 4; ++mt)
#pragma unroll
    for (int r = 0; r < 8; ++r) {
      int m = mt * 16 + ((lane >> 4) << 3) + r;
      float xv = acc[mt][r] + bn;
      float g = 0.5f * xv * (1.f + erff(xv * 0.7071067811865475f));  // exact GELU
      hbuf[(size_t)(row0 + m) * FFD + n] = (bf16)g;
    }
}

// ---------------- launch ------------------------------------------------------

extern "C" void kernel_launch(void* const* d_in, const int* in_sizes, int n_in,
                              void* d_out, int out_size, void* d_ws, size_t ws_size,
                              hipStream_t stream) {
  const float* x    = (const float*)d_in[0];
  const float* pos  = (const float*)d_in[1];
  const float* Wq   = (const float*)d_in[2];
  const float* Wk   = (const float*)d_in[3];
  const float* Wv   = (const float*)d_in[4];
  const float* Wu   = (const float*)d_in[5];
  const float* bu   = (const float*)d_in[6];
  const float* ln1w = (const float*)d_in[7];
  const float* ln1b = (const float*)d_in[8];
  const float* W1   = (const float*)d_in[9];
  const float* b1   = (const float*)d_in[10];
  const float* W2   = (const float*)d_in[11];
  const float* b2   = (const float*)d_in[12];
  const float* ln2w = (const float*)d_in[13];
  const float* ln2b = (const float*)d_in[14];
  const int*   masks= (const int*)  d_in[15];

  char* ws = (char*)d_ws;
  size_t off = 0;
  auto alloc = [&](size_t bytes) { char* p = ws + off; off = (off + bytes + 255) & ~(size_t)255; return p; };
  float* xpe_f = (float*)alloc((size_t)NTOK * EMB * 4);
  bf16*  xpe_b = (bf16*) alloc((size_t)NTOK * EMB * 2);
  bf16*  wq_b  = (bf16*) alloc((size_t)EMB * HE * 2);
  bf16*  wk_b  = (bf16*) alloc((size_t)EMB * HE * 2);
  bf16*  wv_b  = (bf16*) alloc((size_t)EMB * HE * 2);
  bf16*  wuT_b = (bf16*) alloc((size_t)HE * EMB * 2);
  bf16*  w1T_b = (bf16*) alloc((size_t)EMB * FFD * 2);
  bf16*  w2T_b = (bf16*) alloc((size_t)FFD * EMB * 2);
  bf16*  attn_b= (bf16*) alloc((size_t)NTOK * HE * 2);
  float* y1_f  = (float*)alloc((size_t)NTOK * EMB * 4);
  bf16*  y1_b  = (bf16*) alloc((size_t)NTOK * EMB * 2);
  bf16*  h_b   = (bf16*) alloc((size_t)NTOK * FFD * 2);

  // prep
  k_addpos<<<NTOK * EMB / 256, 256, 0, stream>>>(x, pos, xpe_f, xpe_b);
  k_cvt<<<512, 256, 0, stream>>>(Wq, wq_b, EMB * HE);
  k_cvt<<<512, 256, 0, stream>>>(Wk, wk_b, EMB * HE);
  k_cvt<<<512, 256, 0, stream>>>(Wv, wv_b, EMB * HE);
  k_cvtT<<<512, 256, 0, stream>>>(Wu, wuT_b, HE, EMB);   // -> [128][1024]
  k_cvtT<<<256, 256, 0, stream>>>(W1, w1T_b, EMB, FFD);  // -> [512][128]
  k_cvtT<<<256, 256, 0, stream>>>(W2, w2T_b, FFD, EMB);  // -> [128][512]

  // fused attention (259 KB dynamic LDS per workgroup; WGP holds 320 KB)
  (void)hipFuncSetAttribute((const void*)k_attn,
                            hipFuncAttributeMaxDynamicSharedMemorySize, A_LDS);
  k_attn<<<dim3(HEADS, NB), 256, A_LDS, stream>>>(xpe_b, wq_b, wk_b, wv_b, masks, attn_b);

  // Wu + residual + LN1
  k_gemm_ln<<<NTOK / 64, 256, 0, stream>>>(attn_b, HE, wuT_b, bu, xpe_f,
                                           ln1w, ln1b, y1_f, y1_b);
  // FF1 + GELU
  k_ff1<<<dim3(NTOK / 64, FFD / 128), 256, 0, stream>>>(y1_b, w1T_b, b1, h_b);
  // FF2 + residual + LN2 -> out
  k_gemm_ln<<<NTOK / 64, 256, 0, stream>>>(h_b, FFD, w2T_b, b2, y1_f,
                                           ln2w, ln2b, (float*)d_out, (bf16*)nullptr);
}
